// SPP_32985348833517
// MI455X (gfx1250) — matrix operand order repaired and checked by reading, not compiled, not run on hardware
//
#include <hip/hip_runtime.h>
#include <math.h>

typedef __attribute__((ext_vector_type(16))) __bf16 v16bf;
typedef __attribute__((ext_vector_type(8)))  __bf16 v8bf;
typedef __attribute__((ext_vector_type(8)))  float  v8f;
typedef __attribute__((ext_vector_type(4)))  int    v4i;
typedef unsigned short u16;

#define CIN   1024
#define COUT  512
#define NPIX  16384   // B*H*W = 16*32*32
#define HW    1024    // 32*32
#define TM    128
#define TN    64
#define TK    32
#define NK    (CIN / TK)

// ---------------- bf16 hi/lo split helpers ----------------
__device__ __forceinline__ u16 f2bf(float f) {
  unsigned u = __float_as_uint(f);
  u += 0x7FFFu + ((u >> 16) & 1u);      // round-to-nearest-even
  return (u16)(u >> 16);
}
__device__ __forceinline__ float bf2f(u16 h) {
  return __uint_as_float(((unsigned)h) << 16);
}

// ---------------- async global->LDS (gfx1250), guarded probe ----------------
#if __has_builtin(__builtin_amdgcn_global_load_async_to_lds_b128) && \
    __has_builtin(__builtin_amdgcn_s_wait_asynccnt)
#define ASYNC_LDS 1
#define WAIT_ASYNC(n) __builtin_amdgcn_s_wait_asynccnt(n)
typedef v4i __attribute__((address_space(1))) * gptr4;
typedef v4i __attribute__((address_space(3))) * lptr4;
#else
#define ASYNC_LDS 0
#define WAIT_ASYNC(n) ((void)0)
#endif

__device__ __forceinline__ void cp16(const u16* __restrict__ g, u16* l) {
#if ASYNC_LDS
  __builtin_amdgcn_global_load_async_to_lds_b128((gptr4)g, (lptr4)l, 0, 0);
#else
  *(uint4*)l = *(const uint4*)g;
#endif
}

// ---------------- W -> bf16 hi/lo ----------------
__global__ __launch_bounds__(256) void prep_w_kernel(
    const float* __restrict__ W, u16* __restrict__ whi, u16* __restrict__ wlo) {
  int i = blockIdx.x * 256 + threadIdx.x;
  float v = W[i];
  u16 h = f2bf(v);
  whi[i] = h;
  wlo[i] = f2bf(v - bf2f(h));
}

// ---------------- x [b][c][p] f32 -> xT [b*HW + p][c] bf16 hi/lo ----------------
__global__ __launch_bounds__(256) void prep_x_kernel(
    const float* __restrict__ x, u16* __restrict__ xhi, u16* __restrict__ xlo) {
  __shared__ float s[32][33];
  const int b  = blockIdx.z;
  const int c0 = blockIdx.y * 32;
  const int p0 = blockIdx.x * 32;
  const int j  = threadIdx.x & 31;
  const int i0 = threadIdx.x >> 5;   // 0..7
  const float* xb = x + ((size_t)b << 20);
#pragma unroll
  for (int t = 0; t < 4; ++t) {
    int i = i0 + t * 8;
    s[i][j] = xb[(size_t)(c0 + i) * HW + p0 + j];
  }
  __syncthreads();
  u16* hb = xhi + ((size_t)b << 20);
  u16* lb = xlo + ((size_t)b << 20);
#pragma unroll
  for (int t = 0; t < 4; ++t) {
    int i = i0 + t * 8;                       // pixel index within tile
    float v = s[j][i];                        // transposed read (33 pad: no conflicts)
    u16 h = f2bf(v);
    size_t gi = (size_t)(p0 + i) * CIN + c0 + j;
    hb[gi] = h;
    lb[gi] = f2bf(v - bf2f(h));
  }
}

// ---------------- GEMM (bf16x3 WMMA) + BN + Mish, writes y section ----------------
__global__ __launch_bounds__(256) void gemm_bn_mish_kernel(
    const u16* __restrict__ whi, const u16* __restrict__ wlo,
    const u16* __restrict__ xhi, const u16* __restrict__ xlo,
    const float* __restrict__ gamma, const float* __restrict__ beta,
    const float* __restrict__ mean,  const float* __restrict__ var,
    float* __restrict__ out) {
  __shared__ __align__(16) u16 Ah[2][TM][TK];
  __shared__ __align__(16) u16 Al[2][TM][TK];
  __shared__ __align__(16) u16 Bh[2][TN][TK];
  __shared__ __align__(16) u16 Bl[2][TN][TK];

  const int tid  = threadIdx.x;
  const int lane = tid & 31;
  const int wave = tid >> 5;     // 8 waves (wave32)
  const int wm   = wave & 3;     // 4 waves along M
  const int wn   = wave >> 2;    // 2 waves along N
  const int Mbase = blockIdx.y * TM;
  const int Nbase = blockIdx.x * TN;

  v8f acc[2][2];
#pragma unroll
  for (int m = 0; m < 2; ++m)
#pragma unroll
    for (int n = 0; n < 2; ++n)
#pragma unroll
      for (int r = 0; r < 8; ++r) acc[m][n][r] = 0.f;

  auto load_stage = [&](int s) {
    const int buf = s & 1;
    const int kc  = s * TK;
    // A: 128 rows x 32 k (hi+lo). 512 16B chunks each -> 2 per thread.
#pragma unroll
    for (int t = 0; t < 2; ++t) {
      int chunk = tid + t * 256;           // 0..511
      int row   = chunk >> 2;
      int off   = (chunk & 3) * 8;         // elements
      size_t gi = (size_t)(Mbase + row) * CIN + kc + off;
      cp16(whi + gi, &Ah[buf][row][off]);
      cp16(wlo + gi, &Al[buf][row][off]);
    }
    // B: 64 pixel-rows x 32 k (hi+lo). 256 chunks each -> 1 per thread.
    {
      int row = tid >> 2;
      int off = (tid & 3) * 8;
      size_t gi = (size_t)(Nbase + row) * CIN + kc + off;
      cp16(xhi + gi, &Bh[buf][row][off]);
      cp16(xlo + gi, &Bl[buf][row][off]);
    }
  };

  load_stage(0);

  for (int ks = 0; ks < NK; ++ks) {
    const int cur = ks & 1;
    if (ks + 1 < NK) { load_stage(ks + 1); WAIT_ASYNC(6); }  // 6 = next stage in flight
    else             { WAIT_ASYNC(0); }
    __syncthreads();

    const int half = lane >> 4;   // K half-select (ISA A/B layout)
    const int lidx = lane & 15;   // A: M row / B: N column

    v16bf ah[2], al[2], bh[2], bl[2];
#pragma unroll
    for (int m = 0; m < 2; ++m) {
      const u16* ph = &Ah[cur][wm * 32 + m * 16 + lidx][0];
      const u16* pl = &Al[cur][wm * 32 + m * 16 + lidx][0];
      v8bf a0 = *(const v8bf*)(ph + half * 8);
      v8bf a1 = *(const v8bf*)(ph + 16 + half * 8);
      ah[m] = __builtin_shufflevector(a0, a1, 0,1,2,3,4,5,6,7,8,9,10,11,12,13,14,15);
      v8bf l0 = *(const v8bf*)(pl + half * 8);
      v8bf l1 = *(const v8bf*)(pl + 16 + half * 8);
      al[m] = __builtin_shufflevector(l0, l1, 0,1,2,3,4,5,6,7,8,9,10,11,12,13,14,15);
    }
    const int koff = half * 8;
#pragma unroll
    for (int n = 0; n < 2; ++n) {
      const u16* ph = &Bh[cur][wn * 32 + n * 16 + lidx][0];
      const u16* pl = &Bl[cur][wn * 32 + n * 16 + lidx][0];
      v8bf b0 = *(const v8bf*)(ph + koff);
      v8bf b1 = *(const v8bf*)(ph + koff + 16);
      bh[n] = __builtin_shufflevector(b0, b1, 0,1,2,3,4,5,6,7,8,9,10,11,12,13,14,15);
      v8bf c0 = *(const v8bf*)(pl + koff);
      v8bf c1 = *(const v8bf*)(pl + koff + 16);
      bl[n] = __builtin_shufflevector(c0, c1, 0,1,2,3,4,5,6,7,8,9,10,11,12,13,14,15);
    }

#pragma unroll
    for (int m = 0; m < 2; ++m)
#pragma unroll
      for (int n = 0; n < 2; ++n) {
        acc[m][n] = __builtin_amdgcn_wmma_f32_16x16x32_bf16(
            false, ah[m], false, bh[n], (short)0, acc[m][n], false, false);
        acc[m][n] = __builtin_amdgcn_wmma_f32_16x16x32_bf16(
            false, ah[m], false, bl[n], (short)0, acc[m][n], false, false);
        acc[m][n] = __builtin_amdgcn_wmma_f32_16x16x32_bf16(
            false, al[m], false, bh[n], (short)0, acc[m][n], false, false);
      }
    __syncthreads();
  }

  // Epilogue: BN (running stats) + Mish, write into y section (channel offset 1536).
  const int col   = lane & 15;          // N within tile (C/D layout: lane = N)
  const int rbase = (lane >> 4) * 8;    // M row base within 16 (VGPR r = M + half*8)
#pragma unroll
  for (int n = 0; n < 2; ++n) {
    int P  = Nbase + wn * 32 + n * 16 + col;   // global pixel id
    int bi = P >> 10;
    int p  = P & (HW - 1);
#pragma unroll
    for (int m = 0; m < 2; ++m) {
      int ob = Mbase + wm * 32 + m * 16 + rbase;
#pragma unroll
      for (int r = 0; r < 8; ++r) {
        int o = ob + r;
        float inv = gamma[o] * rsqrtf(var[o] + 1e-5f);
        float v   = acc[m][n][r] * inv + (beta[o] - mean[o] * inv);
        float sp  = v > 20.f ? v : log1pf(__expf(v));     // softplus (stable)
        out[(((size_t)(bi * 2048 + 1536 + o)) << 10) + p] = v * tanhf(sp);
      }
    }
  }
}

// ---------------- separable 5x5 max pool (stride 1, same, -inf pad == clipped) ----
__device__ __forceinline__ void pool5(float (*src)[33], float (*dst)[33],
                                      float (*T)[33], int tid) {
#pragma unroll
  for (int t = 0; t < 4; ++t) {
    int pid = tid + t * 256, h = pid >> 5, w = pid & 31;
    int w0 = w - 2 < 0 ? 0 : w - 2;
    int w1 = w + 2 > 31 ? 31 : w + 2;
    float m = src[h][w0];
    for (int ww = w0 + 1; ww <= w1; ++ww) m = fmaxf(m, src[h][ww]);
    T[h][w] = m;
  }
  __syncthreads();
#pragma unroll
  for (int t = 0; t < 4; ++t) {
    int pid = tid + t * 256, h = pid >> 5, w = pid & 31;
    int h0 = h - 2 < 0 ? 0 : h - 2;
    int h1 = h + 2 > 31 ? 31 : h + 2;
    float m = T[h0][w];
    for (int hh = h0 + 1; hh <= h1; ++hh) m = fmaxf(m, T[hh][w]);
    dst[h][w] = m;
  }
  __syncthreads();
}

__device__ __forceinline__ void storeplane(float (*s)[33], float* __restrict__ g, int tid) {
#pragma unroll
  for (int t = 0; t < 4; ++t) {
    int pid = tid + t * 256;
    g[pid] = s[pid >> 5][pid & 31];
  }
}

// P13 = P5∘P5∘P5(y) chain: m5=P5(y), m9=P5(m5), m13=P5(m9) (exact under -inf pad)
__global__ __launch_bounds__(256) void spp_pool_kernel(float* __restrict__ out) {
  __shared__ float A[32][33], Bs[32][33], T[32][33];
  const int plane = blockIdx.x;
  const int b = plane >> 9;
  const int o = plane & 511;
  const size_t base = ((size_t)b * 2048) << 10;
  const float* y = out + base + (((size_t)(1536 + o)) << 10);
  const int tid = threadIdx.x;
#pragma unroll
  for (int t = 0; t < 4; ++t) {
    int pid = tid + t * 256;
    A[pid >> 5][pid & 31] = y[pid];
  }
  __syncthreads();
  pool5(A, Bs, T, tid);                                     // m5
  storeplane(Bs, out + base + (((size_t)(1024 + o)) << 10), tid);
  pool5(Bs, A, T, tid);                                     // m9
  storeplane(A,  out + base + (((size_t)(512 + o)) << 10), tid);
  pool5(A, Bs, T, tid);                                     // m13
  storeplane(Bs, out + base + (((size_t)o) << 10), tid);
}

// ---------------- launcher ----------------
extern "C" void kernel_launch(void* const* d_in, const int* in_sizes, int n_in,
                              void* d_out, int out_size, void* d_ws, size_t ws_size,
                              hipStream_t stream) {
  const float* x     = (const float*)d_in[0];
  const float* W     = (const float*)d_in[1];
  const float* gamma = (const float*)d_in[2];
  const float* beta  = (const float*)d_in[3];
  const float* mean  = (const float*)d_in[4];
  const float* var   = (const float*)d_in[5];
  float* out = (float*)d_out;

  // workspace: Whi|Wlo (1MB each) + xThi|xTlo (32MB each) = 66MB
  u16* whi = (u16*)d_ws;
  u16* wlo = whi + (size_t)COUT * CIN;
  u16* xhi = wlo + (size_t)COUT * CIN;
  u16* xlo = xhi + (size_t)NPIX * CIN;

  prep_w_kernel<<<dim3((COUT * CIN) / 256), 256, 0, stream>>>(W, whi, wlo);
  prep_x_kernel<<<dim3(32, 32, 16), 256, 0, stream>>>(x, xhi, xlo);
  gemm_bn_mish_kernel<<<dim3(NPIX / TN, COUT / TM), 256, 0, stream>>>(
      whi, wlo, xhi, xlo, gamma, beta, mean, var, out);
  spp_pool_kernel<<<dim3(16 * COUT), 256, 0, stream>>>(out);
}
